// SingleHeadAtt_17136919511404
// MI455X (gfx1250) — matrix operand order repaired and checked by reading, not compiled, not run on hardware
//
#include <hip/hip_runtime.h>
#include <hip/hip_bf16.h>

// ---------------------------------------------------------------------------
// Single-head attention for MI455X (gfx1250), wave32 + WMMA bf16.
//   score = (Q K^T)/sqrt(D); attn = softmax(score); ctx = attn V
// d_out = context [B,N,D] f32  ||  attn [B,N,N] f32 (concatenated).
//
// Pipeline (all on `stream`, deterministic, ws rebuilt every call):
//   prep A: K  [B,N,D] f32  -> Kbf [B,N,D]  bf16      (elementwise, d_ws)
//   prep B: V  [B,N,D] f32  -> Vt  [B,D,N]  bf16      (LDS tiled transpose)
//   main:  one workgroup (8 wave32 waves) per 16-query tile:
//     phase 1: S (16x4096) via v_wmma_f32_16x16x32_bf16, bf16 K operands
//              loaded directly (no cvt in inner loop) -> 256 KB LDS strip
//              (only possible with CDNA5's 320 KB/WGP LDS)
//     phase 2: row max/exp/sum, float4 LDS scans + __shfl_xor reductions
//     attn:    coalesced float4 writeback of exp * (1/sum)
//     pack:    in-place f32 -> bf16 packing of the P strip (hazard-free
//              ascending chunks; dst word 2052r+cp < src word 4100r+2cp)
//     phase 3: ctx = P V via WMMA; A = 2x ds_load_b128 (bf16, no cvt),
//              B = 2x global b128 from Vt bf16; 1/sum folded at ctx store
// ---------------------------------------------------------------------------

#define DIMD   128
#define BB     4
#define NN     4096
#define MTILE  16
#define NWAVES 8
#define KEYS_PER_WAVE (NN / NWAVES)     // 512
#define LDS_STRIDE    (NN + 4)          // 4100 floats/row (bank-decorrelating pad)
#define SP_STRIDE     (NN + 8)          // 4104 bf16/row = 2052 words (== 4 mod 64)

typedef __attribute__((ext_vector_type(16))) __bf16 v16bf;
typedef __attribute__((ext_vector_type(8)))  __bf16 v8bf;
typedef __attribute__((ext_vector_type(8)))  float  v8f;
typedef __attribute__((ext_vector_type(4)))  float  f32x4;

__device__ __forceinline__ v16bf cat8(v8bf a, v8bf b) {
    return __builtin_shufflevector(a, b, 0, 1, 2, 3, 4, 5, 6, 7,
                                         8, 9, 10, 11, 12, 13, 14, 15);
}

// Convert 16 staged floats to a bf16 operand vector (lowers to v_cvt_pk_bf16_f32).
__device__ __forceinline__ v16bf pack16f(const float* __restrict__ s) {
    v16bf r;
#pragma unroll
    for (int i = 0; i < 16; ++i) r[i] = (__bf16)s[i];
    return r;
}

// ---------------------------- prep kernels ---------------------------------

// K [B*N*D] f32 -> bf16, 8 elements/thread, b128-friendly.
__global__ __launch_bounds__(256)
void convert_bf16_kernel(const float* __restrict__ in, v8bf* __restrict__ out) {
    const int i = blockIdx.x * 256 + threadIdx.x;          // i < B*N*D/8
    const float* p = in + (size_t)i * 8;
    v8bf o;
#pragma unroll
    for (int j = 0; j < 8; ++j) o[j] = (__bf16)p[j];
    out[i] = o;
}

// V [B][N][D] f32 -> Vt [B][D][N] bf16 via 32x32 LDS tiles.
__global__ __launch_bounds__(256)
void transpose_v_bf16_kernel(const float* __restrict__ V, __bf16* __restrict__ Vt) {
    __shared__ __bf16 tile[32][33];
    const int blk   = blockIdx.x;
    const int tilesPerB = (NN / 32) * (DIMD / 32);
    const int b     = blk / tilesPerB;
    const int r     = blk % tilesPerB;
    const int ntile = r / (DIMD / 32);
    const int dtile = r % (DIMD / 32);
    const int x = threadIdx.x & 31;
    const int y = threadIdx.x >> 5;    // 0..7

    const float* src = V + ((size_t)b * NN + ntile * 32) * DIMD + dtile * 32;
#pragma unroll
    for (int i = 0; i < 4; ++i) {
        const int row = y + i * 8;
        tile[row][x] = (__bf16)src[(size_t)row * DIMD + x];
    }
    __syncthreads();
    __bf16* dst = Vt + ((size_t)b * DIMD + dtile * 32) * NN + ntile * 32;
#pragma unroll
    for (int i = 0; i < 4; ++i) {
        const int row = y + i * 8;
        dst[(size_t)row * NN + x] = tile[x][row];
    }
}

// ---------------------------- main kernel ----------------------------------

__global__ __launch_bounds__(256, 1)
void attn_fused_kernel(const __bf16* __restrict__ Kbf,   // [B][N][D] bf16
                       const float*  __restrict__ Qmat,  // [B][N][D] f32
                       const __bf16* __restrict__ Vt,    // [B][D][N] bf16
                       float* __restrict__ ctx_out,
                       float* __restrict__ attn_out) {
    __shared__ float sS[MTILE * LDS_STRIDE];   // 16 x 4100 f32 = 256.25 KB
    __shared__ float sInv[MTILE];
    __bf16* sP = (__bf16*)sS;                  // packed bf16 P strip (in place)

    const int tid  = threadIdx.x;
    const int wave = tid >> 5;       // 0..7
    const int lane = tid & 31;
    const int lh   = lane >> 4;      // half-wave select (per ISA operand layout)
    const int ll   = lane & 15;

    const int blk   = blockIdx.x;
    const int b     = blk / (NN / MTILE);
    const int qtile = blk % (NN / MTILE);
    const int qbase = qtile * MTILE;

    const float*  __restrict__ Qb  = Qmat + (size_t)b * NN * DIMD;
    const __bf16* __restrict__ Kb  = Kbf  + (size_t)b * NN * DIMD;
    const __bf16* __restrict__ Vtb = Vt   + (size_t)b * DIMD * NN;

    // ---------------- Phase 1: S = scale * Q K^T ---------------------------
    // A operand (Q tile, 16x32 per k-step): converted once, reused by all
    // 128 WMMAs of this wave. 16-bit A layout: lanes 0-15 hold K=0..7&16..23,
    // lanes 16-31 hold K=8..15&24..31.
    v16bf Aop[4];
    {
        const float* qrow = Qb + (size_t)(qbase + ll) * DIMD;
#pragma unroll
        for (int ks = 0; ks < 4; ++ks) {
            float tmp[16];
            const float* p0 = qrow + ks * 32 + lh * 8;
#pragma unroll
            for (int j = 0; j < 8; ++j) tmp[j] = p0[j];
#pragma unroll
            for (int j = 0; j < 8; ++j) tmp[8 + j] = p0[16 + j];
            Aop[ks] = pack16f(tmp);
        }
    }

    const float scale = 0.08838834764831845f;   // 1/sqrt(128)
    const int keyw = wave * KEYS_PER_WAVE;

    for (int nt = 0; nt < KEYS_PER_WAVE / 16; ++nt) {   // 32 key tiles per wave
        const int kb = keyw + nt * 16;
        const __bf16* krow = Kb + (size_t)(kb + ll) * DIMD;
        __builtin_prefetch(krow + 16 * DIMD, 0, 1);     // next tile, same lane row
        v8f acc = {0.f, 0.f, 0.f, 0.f, 0.f, 0.f, 0.f, 0.f};
#pragma unroll
        for (int ks = 0; ks < 4; ++ks) {
            // B operand: lane = key column; two runs of 8 bf16 d-values at 8*lh and 16+8*lh.
            const __bf16* p = krow + ks * 32 + lh * 8;
            v16bf Bop = cat8(*(const v8bf*)p, *(const v8bf*)(p + 16));
            acc = __builtin_amdgcn_wmma_f32_16x16x32_bf16(
                false, Aop[ks], false, Bop, (short)0, acc, false, false);
        }
        // C/D layout: VGPR v -> row v + 8*lh, col = ll within the key tile.
        const int col = kb + ll;
#pragma unroll
        for (int v = 0; v < 8; ++v) {
            const int row = v + lh * 8;
            sS[row * LDS_STRIDE + col] = acc[v] * scale;
        }
    }
    __syncthreads();

    // ---------------- Phase 2: softmax stats (16 lanes per row) ------------
    {
        const int r = tid >> 4;          // row 0..15
        const int c = tid & 15;
        float* row = sS + r * LDS_STRIDE;

        float m = -3.0e38f;
        for (int j4 = c; j4 < NN / 4; j4 += 16) {
            f32x4 v = *(const f32x4*)(row + j4 * 4);
            m = fmaxf(m, fmaxf(fmaxf(v.x, v.y), fmaxf(v.z, v.w)));
        }
#pragma unroll
        for (int off = 8; off > 0; off >>= 1)
            m = fmaxf(m, __shfl_xor(m, off, 32));   // stays within 16-lane group

        float s = 0.f;
        for (int j4 = c; j4 < NN / 4; j4 += 16) {
            f32x4 v = *(const f32x4*)(row + j4 * 4);
            v.x = __expf(v.x - m); v.y = __expf(v.y - m);
            v.z = __expf(v.z - m); v.w = __expf(v.w - m);
            *(f32x4*)(row + j4 * 4) = v;            // keep raw exp; normalize later
            s += (v.x + v.y) + (v.z + v.w);
        }
#pragma unroll
        for (int off = 8; off > 0; off >>= 1)
            s += __shfl_xor(s, off, 32);
        if (c == 0) sInv[r] = 1.0f / s;
    }
    __syncthreads();

    // ---------------- attn writeback: coalesced float4 stores --------------
    {
        float* attn_base = attn_out + ((size_t)b * NN + qbase) * NN;
        for (int idx = tid; idx < MTILE * (NN / 4); idx += 256) {
            const int row = idx / (NN / 4);
            const int c4  = idx % (NN / 4);
            const float inv = sInv[row];
            const float* src = sS + row * LDS_STRIDE + c4 * 4;   // 16B aligned
            f32x4 v = { src[0] * inv, src[1] * inv, src[2] * inv, src[3] * inv };
            *(f32x4*)(attn_base + (size_t)row * NN + c4 * 4) = v;
        }
    }
    __syncthreads();   // all f32 reads of sS complete before in-place packing

    // ---------------- In-place pack: P f32 -> bf16 -------------------------
    // Pair p = (row r, pair-col cp): src words 4100r + 2cp (+1),
    // dst word 2052r + cp.  dst < src for all p > 0, and chunks are processed
    // in ascending order with read->barrier->write->barrier, so no write ever
    // clobbers an unread source.
    {
#pragma unroll 1
        for (int chunk = 0; chunk < 8; ++chunk) {
            const int p0  = chunk * 4096 + tid * 16;    // 16 pairs per thread
            const int row = p0 >> 11;                   // / 2048 pairs-per-row
            const int cp  = p0 & 2047;
            const float* src = sS + row * LDS_STRIDE + cp * 2;
            f32x4 st[8];
#pragma unroll
            for (int j = 0; j < 8; ++j) st[j] = *(const f32x4*)(src + j * 4);
            __syncthreads();
            __bf16* dst = sP + (size_t)row * SP_STRIDE + cp * 2;
#pragma unroll
            for (int j = 0; j < 4; ++j) {
                v8bf pk;
#pragma unroll
                for (int e = 0; e < 8; ++e) pk[e] = (__bf16)st[j * 2 + e / 4][e & 3];
                *(v8bf*)(dst + j * 8) = pk;             // 16B-aligned b128 stores
            }
            __syncthreads();
        }
    }

    // ---------------- Phase 3: ctx = (P V) * inv ---------------------------
    {
        const int dcol = wave * 16;     // this wave's 16 output columns
        const __bf16* vtrow = Vtb + (size_t)(dcol + ll) * NN;
        v8f acc = {0.f, 0.f, 0.f, 0.f, 0.f, 0.f, 0.f, 0.f};

        for (int kb = 0; kb < NN / 32; ++kb) {      // 128 k-steps
            // A: P[16 x 32] bf16 from LDS: two ds_load_b128, zero converts.
            const __bf16* prow = sP + (size_t)ll * SP_STRIDE + kb * 32 + lh * 8;
            v16bf Ap = cat8(*(const v8bf*)prow, *(const v8bf*)(prow + 16));

            // B: V^T row (fixed d-column per lane), two runs of 8 bf16 k-values at 8*lh and 16+8*lh.
            const __bf16* p = vtrow + kb * 32 + lh * 8;
            __builtin_prefetch(p + 32, 0, 1);
            v16bf Bp = cat8(*(const v8bf*)p, *(const v8bf*)(p + 16));

            acc = __builtin_amdgcn_wmma_f32_16x16x32_bf16(
                false, Ap, false, Bp, (short)0, acc, false, false);
        }

        float* cb = ctx_out + ((size_t)b * NN + qbase) * DIMD + dcol + ll;
#pragma unroll
        for (int v = 0; v < 8; ++v) {
            const int row = v + lh * 8;
            cb[(size_t)row * DIMD] = acc[v] * sInv[row];
        }
    }
}

// ---------------------------- host launcher --------------------------------

extern "C" void kernel_launch(void* const* d_in, const int* in_sizes, int n_in,
                              void* d_out, int out_size, void* d_ws, size_t ws_size,
                              hipStream_t stream) {
    (void)in_sizes; (void)n_in; (void)out_size; (void)ws_size;
    // setup_inputs() dict order: key, query, value
    const float* Kmat = (const float*)d_in[0];
    const float* Qmat = (const float*)d_in[1];
    const float* Vmat = (const float*)d_in[2];
    float* ctx  = (float*)d_out;                          // [B,N,D]
    float* attn = ctx + (size_t)BB * NN * DIMD;           // [B,N,N]

    // Workspace: Kbf (4 MB) | Vt bf16 (4 MB); rebuilt every call (deterministic).
    const size_t elems = (size_t)BB * NN * DIMD;          // 2,097,152
    __bf16* Kbf = (__bf16*)d_ws;
    __bf16* Vt  = Kbf + elems;

    convert_bf16_kernel<<<dim3((unsigned)(elems / 8 / 256)), dim3(256), 0, stream>>>(
        Kmat, (v8bf*)Kbf);
    transpose_v_bf16_kernel<<<dim3(BB * (NN / 32) * (DIMD / 32)), dim3(256), 0, stream>>>(
        Vmat, Vt);
    attn_fused_kernel<<<dim3(BB * (NN / MTILE)), dim3(256), 0, stream>>>(
        Kbf, Qmat, Vt, ctx, attn);
}
